// LSTMAggregator_27376121545087
// MI455X (gfx1250) — matrix operand order repaired and checked by reading, not compiled, not run on hardware
//
#include <hip/hip_runtime.h>

// ---------------------------------------------------------------------------
// Ragged LSTM final-hidden-state, MI455X (gfx1250, wave32, WMMA).
//
//   * W_ih / W_hh -> bf16 once (prep kernel), biases combined.
//   * Persistent main kernel: each 256-thread block owns BT=32 batch rows for
//     all 32 timesteps -> 313 blocks (fills 256+ WGPs, multiple co-resident
//     blocks per WGP thanks to the ~52 KB LDS footprint).
//   * h (bf16, double-buffered) + per-step x tile (bf16) live in LDS;
//     the cell state c lives entirely in REGISTERS (each wave exclusively
//     owns its (row, hidden-slice) elements for the whole sequence).
//   * Per step, gates[32,1024] = x_t @ W_ih^T + h @ W_hh^T + b via
//     v_wmma_f32_16x16x32_bf16; both products share one f32 accumulator,
//     each weight fragment read from L2 exactly once per block-step.
//   * Each wave owns two 16-wide hidden slices and computes all 4 gate tiles
//     (i,f,g,o) for them -> sigmoid/tanh epilogue + c/h update stay local.
//   * global_prefetch of next step's x rows overlaps the WMMA phase.
//   * Ragged lengths: freeze rows past their length; d_out gets h at the
//     last valid step (last valid write wins == h at t = len-1).
// ---------------------------------------------------------------------------

typedef unsigned short ushort_t;
typedef __attribute__((ext_vector_type(16))) __bf16       v16bf;
typedef __attribute__((ext_vector_type(8)))  float        v8f;
typedef __attribute__((ext_vector_type(8)))  unsigned int v8u;

#define N_SEQ 10000
#define L_SEQ 32
#define D_HID 256
#define G_DIM 1024   // 4*D
#define BT    32     // batch rows per block
#define NTHREADS 256

__device__ __forceinline__ ushort_t f2bf(float f) {
  unsigned int u = __float_as_uint(f);
  u += 0x7FFFu + ((u >> 16) & 1u);   // round-to-nearest-even
  return (ushort_t)(u >> 16);
}
__device__ __forceinline__ float bf2f(ushort_t b) {
  return __uint_as_float(((unsigned int)b) << 16);
}
__device__ __forceinline__ float sigf(float v) {
  return 1.0f / (1.0f + __expf(-v));
}
__device__ __forceinline__ float tanh_fast(float v) {
  return 1.0f - 2.0f / (__expf(2.0f * v) + 1.0f);
}

// Assemble a 32-byte WMMA fragment from two 16-byte chunks.
__device__ __forceinline__ v16bf make_frag(const ushort_t* p0, const ushort_t* p1) {
  uint4 a = *(const uint4*)p0;
  uint4 b = *(const uint4*)p1;
  v8u r;
  r[0] = a.x; r[1] = a.y; r[2] = a.z; r[3] = a.w;
  r[4] = b.x; r[5] = b.y; r[6] = b.z; r[7] = b.w;
  return __builtin_bit_cast(v16bf, r);
}

// ---------------------------------------------------------------------------
// Prep: fp32 weights -> bf16, combine biases.
// ---------------------------------------------------------------------------
__global__ void lstm_prep_kernel(const float* __restrict__ Wih,
                                 const float* __restrict__ Whh,
                                 const float* __restrict__ bih,
                                 const float* __restrict__ bhh,
                                 ushort_t* __restrict__ wih_bf,
                                 ushort_t* __restrict__ whh_bf,
                                 float* __restrict__ bias) {
  int idx = blockIdx.x * blockDim.x + threadIdx.x;
  if (idx < G_DIM * D_HID) {
    wih_bf[idx] = f2bf(Wih[idx]);
    whh_bf[idx] = f2bf(Whh[idx]);
  }
  if (idx < G_DIM) bias[idx] = bih[idx] + bhh[idx];
}

// ---------------------------------------------------------------------------
// Main persistent recurrent kernel.
// ---------------------------------------------------------------------------
__launch_bounds__(NTHREADS, 1)
__global__ void lstm_step_kernel(const float* __restrict__ x,
                                 const int* __restrict__ lengths,
                                 const ushort_t* __restrict__ wih,
                                 const ushort_t* __restrict__ whh,
                                 const float* __restrict__ bias,
                                 float* __restrict__ out) {
  __shared__ alignas(16) ushort_t h_buf[2][BT * D_HID];  // 2 x 16 KB, bf16
  __shared__ alignas(16) ushort_t x_s[BT * D_HID];       // 16 KB, bf16
  __shared__ float b_s[G_DIM];                           // 4 KB
  __shared__ int   len_s[BT];

  const int tid     = threadIdx.x;
  const int lane    = tid & 31;
  const int wave    = tid >> 5;      // 0..7
  const int lanelo  = lane & 15;     // col within tile / row within A tile
  const int halfSel = lane >> 4;     // 0: lanes 0-15, 1: lanes 16-31
  const int n0      = blockIdx.x * BT;

  // Cell state in registers: [jj][mtile][r] for this wave's owned elements.
  float c_reg[2][2][8];
#pragma unroll
  for (int a = 0; a < 2; ++a)
#pragma unroll
    for (int b = 0; b < 2; ++b)
#pragma unroll
      for (int r = 0; r < 8; ++r) c_reg[a][b][r] = 0.0f;

  // ---- init state + stage constants ----
  for (int i = tid; i < BT * D_HID; i += NTHREADS) h_buf[0][i] = 0;
  for (int i = tid; i < G_DIM; i += NTHREADS) b_s[i] = bias[i];
  if (tid < BT) {
    int g = n0 + tid;
    len_s[tid] = (g < N_SEQ) ? lengths[g] : 0;   // OOB rows never valid
  }
  __syncthreads();

  for (int t = 0; t < L_SEQ; ++t) {
    const int curb = t & 1, nxtb = curb ^ 1;

    // ---- stage x[:, t, :] tile -> bf16 in LDS (coalesced float4) ----
    for (int i = tid; i < (BT * D_HID) / 4; i += NTHREADS) {
      int flat = i * 4;
      int r = flat >> 8;       // row 0..31
      int d = flat & 255;      // col 0..255
      int gRow = n0 + r;
      float4 v = make_float4(0.f, 0.f, 0.f, 0.f);
      if (gRow < N_SEQ)
        v = *(const float4*)&x[(size_t)gRow * (L_SEQ * D_HID) + t * D_HID + d];
      ushort_t* dst = &x_s[r * D_HID + d];
      dst[0] = f2bf(v.x); dst[1] = f2bf(v.y);
      dst[2] = f2bf(v.z); dst[3] = f2bf(v.w);
    }
    __syncthreads();

    // ---- prefetch next step's x rows (overlaps WMMA phase) ----
    if (t + 1 < L_SEQ) {
      // 32 rows x 1 KB = 32 KB; 256 threads -> one 128B line each
      int r = tid >> 3;                     // row 0..31
      int seg = (tid & 7) * 128;            // byte segment within the row
      int gRow = n0 + r;
      if (gRow < N_SEQ) {
        const char* p = (const char*)&x[(size_t)gRow * (L_SEQ * D_HID) +
                                        (t + 1) * D_HID];
        __builtin_prefetch(p + seg, 0, 1);
      }
    }

    // ---- GEMM: wave owns hidden slices j = wave and wave+8 ----
#pragma unroll
    for (int jj = 0; jj < 2; ++jj) {
      const int j = wave + jj * 8;
      const int dcolBase = j * 16;
      const int gcolLane = dcolBase + lanelo;          // B-matrix column

      v8f acc[2][4] = {};                              // [mtile][gate]

      for (int k = 0; k < 8; ++k) {                    // K = 256 in chunks of 32
        const int kb = k * 32;
        v16bf Bx[4], Bh[4];
#pragma unroll
        for (int g = 0; g < 4; ++g) {
          // B[k][n] = W[gate_col][k]; lane loads 16 contiguous K (32 B)
          const ushort_t* px = wih + (size_t)(g * D_HID + gcolLane) * D_HID
                                   + kb + halfSel * 16;
          const ushort_t* ph = whh + (size_t)(g * D_HID + gcolLane) * D_HID
                                   + kb + halfSel * 16;
          Bx[g] = make_frag(px, px + 8);
          Bh[g] = make_frag(ph, ph + 8);
        }
#pragma unroll
        for (int mi = 0; mi < 2; ++mi) {
          const int rowL = mi * 16 + lanelo;           // A row = lane%16
          const ushort_t* xp = &x_s[rowL * D_HID + kb];
          const ushort_t* hp = &h_buf[curb][rowL * D_HID + kb];
          // A layout: elems 0-15 = K(halfSel*16..+16), contiguous
          v16bf Ax = make_frag(xp + halfSel * 16, xp + halfSel * 16 + 8);
          v16bf Ah = make_frag(hp + halfSel * 16, hp + halfSel * 16 + 8);
#pragma unroll
          for (int g = 0; g < 4; ++g) {
            acc[mi][g] = __builtin_amdgcn_wmma_f32_16x16x32_bf16(
                false, Ax, false, Bx[g], (short)0, acc[mi][g], false, false);
            acc[mi][g] = __builtin_amdgcn_wmma_f32_16x16x32_bf16(
                false, Ah, false, Bh[g], (short)0, acc[mi][g], false, false);
          }
        }
      }

      // ---- nonlinear epilogue + state update (c stays in registers) ----
      const float bI = b_s[0 * D_HID + dcolBase + lanelo];
      const float bF = b_s[1 * D_HID + dcolBase + lanelo];
      const float bG = b_s[2 * D_HID + dcolBase + lanelo];
      const float bO = b_s[3 * D_HID + dcolBase + lanelo];
#pragma unroll
      for (int mi = 0; mi < 2; ++mi) {
        const int mBase = mi * 16;
#pragma unroll
        for (int r = 0; r < 8; ++r) {
          // C layout: row = r + 8*halfSel, col = lane%16
          const int rowL = mBase + r + halfSel * 8;
          const int dcol = dcolBase + lanelo;
          const int idx  = rowL * D_HID + dcol;
          const float iv = sigf(acc[mi][0][r] + bI);
          const float fv = sigf(acc[mi][1][r] + bF);
          const float gv = tanh_fast(acc[mi][2][r] + bG);
          const float ov = sigf(acc[mi][3][r] + bO);
          const float cn = fv * c_reg[jj][mi][r] + iv * gv;
          const float hn = ov * tanh_fast(cn);
          const bool valid = (t < len_s[rowL]);
          const float hw = valid ? hn : bf2f(h_buf[curb][idx]);
          h_buf[nxtb][idx] = f2bf(hw);              // always carry h forward
          if (valid) {
            c_reg[jj][mi][r] = cn;
            out[(size_t)(n0 + rowL) * D_HID + dcol] = hn;  // last valid wins
          }
        }
      }
    }
    __syncthreads();   // x_s / h_buf reuse fence for next step
  }
}

// ---------------------------------------------------------------------------
extern "C" void kernel_launch(void* const* d_in, const int* in_sizes, int n_in,
                              void* d_out, int out_size, void* d_ws, size_t ws_size,
                              hipStream_t stream) {
  const float* x    = (const float*)d_in[0];
  const int*   len  = (const int*)d_in[1];
  const float* Wih  = (const float*)d_in[2];
  const float* Whh  = (const float*)d_in[3];
  const float* bih  = (const float*)d_in[4];
  const float* bhh  = (const float*)d_in[5];
  float*       out  = (float*)d_out;

  ushort_t* wih_bf = (ushort_t*)d_ws;
  ushort_t* whh_bf = wih_bf + (size_t)G_DIM * D_HID;
  float*    bias   = (float*)(whh_bf + (size_t)G_DIM * D_HID);

  // Prep: 262144 threads cover both weight matrices + bias.
  lstm_prep_kernel<<<(G_DIM * D_HID + NTHREADS - 1) / NTHREADS, NTHREADS, 0, stream>>>(
      Wih, Whh, bih, bhh, wih_bf, whh_bf, bias);

  const int nblocks = (N_SEQ + BT - 1) / BT;  // 313
  lstm_step_kernel<<<nblocks, NTHREADS, 0, stream>>>(
      x, len, wih_bf, whh_bf, bias, out);
}
